// mLSTMCell_89807766159689
// MI455X (gfx1250) — matrix operand order repaired and checked by reading, not compiled, not run on hardware
//
#include <hip/hip_runtime.h>
#include <hip/hip_bf16.h>
#include <math.h>

// ---------------------------------------------------------------------------
// mLSTM parallel forward for MI455X (gfx1250), wave32 + WMMA bf16.
//
// Shapes: B=1, S=2048, D=1024, NH=8, DH=128.
//
// Math rewrite: with c_i = cumsum(log_sigmoid(fg))_i, g_j = ig_j - c_j,
// M_i = running_max_{j<=i} g_j:
//   D_mat[i,j]       = exp(g_j - M_i)            (j <= i, else 0)
//   exp(-max_logD_i) = exp(-(c_i + M_i))
// so the S x S stage is a causal tiled matmul with a separable per-element
// weight -- no online rescaling needed.  1/sqrt(DH) is folded into the
// exponent: C[i,j] = qk[i,j] * exp(g_j + ln(1/sqrt(DH)) - M_i).
//
// v4: branchless diagonal masking + uniform interior fast path; keeps the
// v3 double-buffered LDS staging (next-tile b128 global loads fly under the
// current tile's 16 WMMAs; one barrier per j-step).
// ---------------------------------------------------------------------------

#define SLEN   2048
#define DMODEL 1024
#define NHEAD  8
#define DHEAD  128

typedef __attribute__((ext_vector_type(16))) __bf16 v16bf;
typedef __attribute__((ext_vector_type(8)))  __bf16 v8bf;
typedef __attribute__((ext_vector_type(4)))  __bf16 v4bf;
typedef __attribute__((ext_vector_type(8)))  float  v8f;

static __device__ __forceinline__ v16bf concat8(v8bf lo, v8bf hi) {
    return __builtin_shufflevector(lo, hi, 0, 1, 2, 3, 4, 5, 6, 7,
                                   8, 9, 10, 11, 12, 13, 14, 15);
}

// ---------------------------------------------------------------------------
// Kernel 1: gate pre-activations.  One block per sequence position, qkv row
// staged in LDS, wave w computes head w's two 3072-length dot products.
// Writes ig and lf = log_sigmoid(fg).
// ---------------------------------------------------------------------------
__global__ __launch_bounds__(256) void mlstm_gates_kernel(
    const float* __restrict__ q, const float* __restrict__ k,
    const float* __restrict__ v,
    const float* __restrict__ igw, const float* __restrict__ igb,
    const float* __restrict__ fgw, const float* __restrict__ fgb,
    float* __restrict__ ws_ig, float* __restrict__ ws_lf)
{
    __shared__ float sm[3 * DMODEL];
    const int s = blockIdx.x;
    for (int i = threadIdx.x; i < DMODEL; i += 256) {
        sm[i]              = q[s * DMODEL + i];
        sm[DMODEL + i]     = k[s * DMODEL + i];
        sm[2 * DMODEL + i] = v[s * DMODEL + i];
    }
    __syncthreads();

    const int head = threadIdx.x >> 5;   // 8 waves == 8 heads
    const int lane = threadIdx.x & 31;
    const float* wi = igw + head * (3 * DMODEL);
    const float* wf = fgw + head * (3 * DMODEL);

    float si = 0.f, sf = 0.f;
    for (int i = lane; i < 3 * DMODEL; i += 32) {
        float x = sm[i];
        si = fmaf(x, wi[i], si);
        sf = fmaf(x, wf[i], sf);
    }
#pragma unroll
    for (int m = 16; m >= 1; m >>= 1) {
        si += __shfl_xor(si, m);
        sf += __shfl_xor(sf, m);
    }
    if (lane == 0) {
        float ig = si + igb[head];
        float fg = sf + fgb[head];
        // stable log_sigmoid
        float lf = fminf(fg, 0.f) - log1pf(__expf(-fabsf(fg)));
        ws_ig[head * SLEN + s] = ig;
        ws_lf[head * SLEN + s] = lf;
    }
}

// ---------------------------------------------------------------------------
// Kernel 2: per-head scans.  Single block, wave h scans head h:
//   c = inclusive cumsum(lf); g = ig - c; M = inclusive runmax(g)
// ---------------------------------------------------------------------------
__global__ __launch_bounds__(256) void mlstm_scan_kernel(
    const float* __restrict__ ws_ig, const float* __restrict__ ws_lf,
    float* __restrict__ ws_c, float* __restrict__ ws_g,
    float* __restrict__ ws_M)
{
    const int head = threadIdx.x >> 5;
    const int lane = threadIdx.x & 31;
    float ccarry = 0.f;
    float mcarry = -INFINITY;
    for (int t0 = 0; t0 < SLEN; t0 += 32) {
        const int idx = head * SLEN + t0 + lane;
        float x = ws_lf[idx];
#pragma unroll
        for (int d = 1; d < 32; d <<= 1) {
            float y = __shfl_up(x, d);
            if (lane >= d) x += y;
        }
        float c = ccarry + x;
        ccarry  = __shfl(c, 31);

        float g  = ws_ig[idx] - c;
        float mg = g;
#pragma unroll
        for (int d = 1; d < 32; d <<= 1) {
            float y = __shfl_up(mg, d);
            if (lane >= d) mg = fmaxf(mg, y);
        }
        float M = fmaxf(mcarry, mg);
        mcarry  = __shfl(M, 31);

        ws_c[idx] = c;
        ws_g[idx] = g;
        ws_M[idx] = M;
    }
}

// ---------------------------------------------------------------------------
// Kernel 3: causal weighted attention + normalizer + fused per-head LayerNorm.
// grid = (S/128, NH); 8 waves/block; wave owns a 16-row strip of one head.
// All waves sweep the block-wide causal column range; K/V tiles are staged
// once per workgroup in double-buffered LDS (bf16) and shared 8x.
//
// Fragment layouts (CDNA5 ISA 7.12.2, wave32):
//   A (16x32 bf16): lane%16 = row; K pairs: (lane>=16?8:0)+(v&3)*2+(v>>2)*16
//   B (32x16 bf16): lane%16 = col;  K pairs as A: (lane>=16?8:0)+(v&3)*2+(v>>2)*16
//   C/D (16x16 f32): VGPR r -> row r + (lane>=16?8:0), col = lane%16
// ---------------------------------------------------------------------------
__global__ __launch_bounds__(256) void mlstm_attn_kernel(
    const float* __restrict__ q, const float* __restrict__ k,
    const float* __restrict__ v,
    const float* __restrict__ ws_c, const float* __restrict__ ws_g,
    const float* __restrict__ ws_M,
    const float* __restrict__ norm_w, float* __restrict__ out)
{
    // Double-buffered staging tiles (bf16) + per-wave P staging.
    __shared__ __align__(16) __bf16 lds_k[2][32][DHEAD];    // K rows       (16 KB)
    __shared__ __align__(16) __bf16 lds_vT[2][DHEAD][32];   // V transposed (16 KB)
    __shared__ __align__(16) __bf16 lds_p[8][16][32];       // P tiles      ( 8 KB)
    __shared__ __align__(16) float  lds_g[2][32];           // g_j per tile

    const int tid    = threadIdx.x;
    const int head   = blockIdx.y;
    const int wave   = tid >> 5;
    const int lane   = tid & 31;
    const int lane16 = lane & 15;
    const bool hi    = lane >= 16;
    const int hioff  = hi ? 8 : 0;
    const int row0   = blockIdx.x * 128 + wave * 16;
    const int rowmax = row0 + 15;

    const float* qh = q + head * DHEAD;
    const float* kh = k + head * DHEAD;
    const float* vh = v + head * DHEAD;
    const float* ch = ws_c + head * SLEN;
    const float* gh = ws_g + head * SLEN;
    const float* Mh = ws_M + head * SLEN;

    // ---- Q strip as 4 bf16 A-fragments (K chunks of 32 along DH=128) ----
    v16bf qa[4];
    {
        const float* qrow = qh + (row0 + lane16) * DMODEL;
#pragma unroll
        for (int kc = 0; kc < 4; ++kc) {
#pragma unroll
            for (int vv = 0; vv < 8; ++vv) {
                int kd = kc * 32 + (hi ? 8 : 0) + (vv & 3) * 2 + (vv >> 2) * 16;
                float2 f = *(const float2*)(qrow + kd);
                qa[kc][2 * vv]     = (__bf16)f.x;
                qa[kc][2 * vv + 1] = (__bf16)f.y;
            }
        }
    }

    // ---- per-row M_i and c_i broadcast into registers indexed by r ----
    float mrow[8], crow[8];
    {
        float ml = Mh[row0 + lane16];
        float cl = ch[row0 + lane16];
#pragma unroll
        for (int r = 0; r < 8; ++r) {
            int src = hi ? (24 + r) : r;   // lane holding row (row0 + r + 8*hi)
            mrow[r] = __shfl(ml, src);
            crow[r] = __shfl(cl, src);
        }
    }

    v8f cacc[8];
#pragma unroll
    for (int t = 0; t < 8; ++t) cacc[t] = (v8f){};
    float rowsum[8] = {0.f, 0.f, 0.f, 0.f, 0.f, 0.f, 0.f, 0.f};

    const float ln_inv_s  = -2.4260151319598084f;  // ln(1/sqrt(128))
    const int   ncols_blk = blockIdx.x * 128 + 128; // block-uniform causal bound

    // staging task coordinates (constant per thread)
    const int kj0 = tid >> 5;            // K: j row of task 0 (0..7)
    const int dh4 = (tid & 31) << 2;     // dh group 0,4,...,124
    const int vp0 = tid >> 5;            // V: j-pair of task 0 (0..7)

    // ==== prologue: stage tile j0=0 into buffer 0 ====
    {
#pragma unroll
        for (int i = 0; i < 4; ++i) {
            int jrel = kj0 + 8 * i;
            float4 f = *(const float4*)(kh + jrel * DMODEL + dh4);
            v4bf c4;
            c4[0] = (__bf16)f.x; c4[1] = (__bf16)f.y;
            c4[2] = (__bf16)f.z; c4[3] = (__bf16)f.w;
            *(v4bf*)&lds_k[0][jrel][dh4] = c4;
        }
#pragma unroll
        for (int i = 0; i < 2; ++i) {
            int p = vp0 + 8 * i;
            const float* vrow = vh + (2 * p) * DMODEL + dh4;
            float4 a = *(const float4*)vrow;
            float4 b = *(const float4*)(vrow + DMODEL);
            lds_vT[0][dh4 + 0][2 * p] = (__bf16)a.x; lds_vT[0][dh4 + 0][2 * p + 1] = (__bf16)b.x;
            lds_vT[0][dh4 + 1][2 * p] = (__bf16)a.y; lds_vT[0][dh4 + 1][2 * p + 1] = (__bf16)b.y;
            lds_vT[0][dh4 + 2][2 * p] = (__bf16)a.z; lds_vT[0][dh4 + 2][2 * p + 1] = (__bf16)b.z;
            lds_vT[0][dh4 + 3][2 * p] = (__bf16)a.w; lds_vT[0][dh4 + 3][2 * p + 1] = (__bf16)b.w;
        }
        if (tid < 32) lds_g[0][tid] = gh[tid];
    }
    __syncthreads();

    int cur = 0;
    for (int j0 = 0; j0 < ncols_blk; j0 += 32, cur ^= 1) {
        const int  jn        = j0 + 32;
        const bool have_next = jn < ncols_blk;
        const int  nxt       = cur ^ 1;

        // ==== issue next-tile global loads (kept in VGPRs under compute) ====
        float4 knxt[4];
        float4 vna[2], vnb[2];
        float  gnxt = 0.f;
        if (have_next) {
#pragma unroll
            for (int i = 0; i < 4; ++i)
                knxt[i] = *(const float4*)(kh + (jn + kj0 + 8 * i) * DMODEL + dh4);
#pragma unroll
            for (int i = 0; i < 2; ++i) {
                const float* vrow = vh + (jn + 2 * (vp0 + 8 * i)) * DMODEL + dh4;
                vna[i] = *(const float4*)vrow;
                vnb[i] = *(const float4*)(vrow + DMODEL);
            }
            if (tid < 32) gnxt = gh[jn + tid];
        }

        // ==== compute from buffer `cur` (waves past causal bound skip) ====
        if (j0 <= rowmax) {
            // two 16x16 score tiles -> weighted P halves in per-wave LDS
#pragma unroll
            for (int t2 = 0; t2 < 2; ++t2) {
                const int jt = j0 + t2 * 16;
                if (jt <= rowmax) {
                    const int jrel = t2 * 16 + lane16;   // column within tile
                    v8f sacc = (v8f){};
#pragma unroll
                    for (int kc = 0; kc < 4; ++kc) {
                        const int base = kc * 32 + (hi ? 8 : 0);
                        v8bf lo  = *(const v8bf*)&lds_k[cur][jrel][base];
                        v8bf hi8 = *(const v8bf*)&lds_k[cur][jrel][base + 16];
                        sacc = __builtin_amdgcn_wmma_f32_16x16x32_bf16(
                            false, qa[kc], false, concat8(lo, hi8), (short)0,
                            sacc, false, false);
                    }
                    // 1/sqrt(DH) folded into the exponent
                    const float gcol = lds_g[cur][jrel] + ln_inv_s;
                    if (jt + 15 <= row0) {
                        // interior tile: fully unmasked fast path
#pragma unroll
                        for (int r = 0; r < 8; ++r) {
                            float p = sacc[r] * __expf(gcol - mrow[r]);
                            rowsum[r] += p;
                            lds_p[wave][r + hioff][jrel] = (__bf16)p;
                        }
                    } else {
                        // diagonal tile: branchless select per element
                        const int jcol = j0 + jrel;
#pragma unroll
                        for (int r = 0; r < 8; ++r) {
                            float e = sacc[r] * __expf(gcol - mrow[r]);
                            float p = (jcol <= row0 + r + hioff) ? e : 0.f;
                            rowsum[r] += p;
                            lds_p[wave][r + hioff][jrel] = (__bf16)p;
                        }
                    }
                } else {
#pragma unroll
                    for (int r = 0; r < 8; ++r)
                        lds_p[wave][r + hioff][t2 * 16 + lane16] = (__bf16)0.f;
                }
            }

            // in-wave cross-lane LDS dep: drain DS stores before reload
            asm volatile("s_wait_dscnt 0" ::: "memory");

            // reload P tile (16x32) in A-fragment layout: 2x b128 DS
            v16bf pa;
            {
                const int base = hi ? 8 : 0;
                v8bf lo  = *(const v8bf*)&lds_p[wave][lane16][base];
                v8bf hi8 = *(const v8bf*)&lds_p[wave][lane16][base + 16];
                pa = concat8(lo, hi8);
            }

            // P @ V: 8 WMMAs cover DH=128 output columns
#pragma unroll
            for (int t = 0; t < 8; ++t) {
                const int dh   = t * 16 + lane16;
                const int base = hi ? 8 : 0;
                v8bf lo  = *(const v8bf*)&lds_vT[cur][dh][base];
                v8bf hi8 = *(const v8bf*)&lds_vT[cur][dh][base + 16];
                cacc[t] = __builtin_amdgcn_wmma_f32_16x16x32_bf16(
                    false, pa, false, concat8(lo, hi8), (short)0, cacc[t],
                    false, false);
            }
        }

        // ==== convert + store next tile into buffer `nxt` ====
        if (have_next) {
#pragma unroll
            for (int i = 0; i < 4; ++i) {
                v4bf c4;
                c4[0] = (__bf16)knxt[i].x; c4[1] = (__bf16)knxt[i].y;
                c4[2] = (__bf16)knxt[i].z; c4[3] = (__bf16)knxt[i].w;
                *(v4bf*)&lds_k[nxt][kj0 + 8 * i][dh4] = c4;
            }
#pragma unroll
            for (int i = 0; i < 2; ++i) {
                int p = vp0 + 8 * i;
                lds_vT[nxt][dh4 + 0][2 * p] = (__bf16)vna[i].x; lds_vT[nxt][dh4 + 0][2 * p + 1] = (__bf16)vnb[i].x;
                lds_vT[nxt][dh4 + 1][2 * p] = (__bf16)vna[i].y; lds_vT[nxt][dh4 + 1][2 * p + 1] = (__bf16)vnb[i].y;
                lds_vT[nxt][dh4 + 2][2 * p] = (__bf16)vna[i].z; lds_vT[nxt][dh4 + 2][2 * p + 1] = (__bf16)vnb[i].z;
                lds_vT[nxt][dh4 + 3][2 * p] = (__bf16)vna[i].w; lds_vT[nxt][dh4 + 3][2 * p + 1] = (__bf16)vnb[i].w;
            }
            if (tid < 32) lds_g[nxt][tid] = gnxt;
        }

        __syncthreads();   // cur reads done everywhere; nxt stores visible
    }

    // ---- finalize: normalizer + fused per-head LayerNorm + store ----
#pragma unroll
    for (int r = 0; r < 8; ++r) {
        float rs = rowsum[r];
#pragma unroll
        for (int m = 1; m < 16; m <<= 1) rs += __shfl_xor(rs, m);
        float norm = fmaxf(fabsf(rs), __expf(-(crow[r] + mrow[r]))) + 1e-6f;
        float inv  = 1.f / norm;

        float xs[8];
        float mean = 0.f;
#pragma unroll
        for (int t = 0; t < 8; ++t) {
            float x = cacc[t][r] * inv;
            xs[t] = x;
            mean += x;
        }
#pragma unroll
        for (int m = 1; m < 16; m <<= 1) mean += __shfl_xor(mean, m);
        mean *= (1.f / 128.f);

        float m2 = 0.f;
#pragma unroll
        for (int t = 0; t < 8; ++t) {
            float d = xs[t] - mean;
            m2 = fmaf(d, d, m2);
        }
#pragma unroll
        for (int m = 1; m < 16; m <<= 1) m2 += __shfl_xor(m2, m);
        float rstd = rsqrtf(m2 * (1.f / 128.f) + 1e-5f);

        const int row = row0 + r + hioff;
        float* orow = out + row * DMODEL + head * DHEAD;
#pragma unroll
        for (int t = 0; t < 8; ++t) {
            float nw = norm_w[head * DHEAD + t * 16 + lane16];
            orow[t * 16 + lane16] = (xs[t] - mean) * rstd * nw;
        }
    }
}

// ---------------------------------------------------------------------------
// Host launcher
// ---------------------------------------------------------------------------
extern "C" void kernel_launch(void* const* d_in, const int* in_sizes, int n_in,
                              void* d_out, int out_size, void* d_ws,
                              size_t ws_size, hipStream_t stream)
{
    const float* q   = (const float*)d_in[0];
    const float* k   = (const float*)d_in[1];
    const float* v   = (const float*)d_in[2];
    const float* igw = (const float*)d_in[3];
    const float* igb = (const float*)d_in[4];
    const float* fgw = (const float*)d_in[5];
    const float* fgb = (const float*)d_in[6];
    const float* nw  = (const float*)d_in[7];
    float* out = (float*)d_out;

    // workspace: 5 arrays of NH*S f32 (64 KB each)
    float* ws    = (float*)d_ws;
    float* ws_ig = ws;
    float* ws_lf = ws + 1 * NHEAD * SLEN;
    float* ws_c  = ws + 2 * NHEAD * SLEN;
    float* ws_g  = ws + 3 * NHEAD * SLEN;
    float* ws_M  = ws + 4 * NHEAD * SLEN;

    mlstm_gates_kernel<<<SLEN, 256, 0, stream>>>(q, k, v, igw, igb, fgw, fgb,
                                                 ws_ig, ws_lf);
    mlstm_scan_kernel<<<1, 256, 0, stream>>>(ws_ig, ws_lf, ws_c, ws_g, ws_M);
    mlstm_attn_kernel<<<dim3(SLEN / 128, NHEAD), 256, 0, stream>>>(
        q, k, v, ws_c, ws_g, ws_M, nw, out);
}
